// SwinTransformer_738734375475
// MI455X (gfx1250) — matrix operand order repaired and checked by reading, not compiled, not run on hardware
//
#include <hip/hip_runtime.h>

// ---------------------------------------------------------------------------
// CDNA5 (gfx1250) Swin block: bf16 WMMA everywhere, fp32 accumulate.
// Native __bf16 converts + async global->LDS tile copies (ASYNCcnt).
// ---------------------------------------------------------------------------

typedef __attribute__((ext_vector_type(16))) __bf16 v16bf;
typedef __attribute__((ext_vector_type(8)))  float  v8f;

union BF16Frag { v16bf v; __bf16 s[16]; uint4 q[2]; };

#define DEVI __device__ __forceinline__

DEVI __bf16 tobf(float f) { return (__bf16)f; }   // native v_cvt RTNE

DEVI v8f splat8(float b) { v8f r = {b, b, b, b, b, b, b, b}; return r; }

DEVI v8f wmma_bf16(v16bf a, v16bf b, v8f c) {
    return __builtin_amdgcn_wmma_f32_16x16x32_bf16(false, a, false, b,
                                                   (short)0, c, false, false);
}

// A fragment: 16x32 (MxK) from row-major bf16 source.
// Layout used here: lane<16 hold K 0..15, lane>=16 hold K 16..31 (contiguous per lane).
DEVI v16bf frag_a(const __bf16* base, int ld, int m0, int k0, int lane) {
    BF16Frag f;
    const __bf16* row =
        base + (size_t)(m0 + (lane & 15)) * ld + k0 + ((lane >> 4) << 4);
    f.q[0] = *(const uint4*)(row);        // K +0..7
    f.q[1] = *(const uint4*)(row + 8);    // K +8..15
    return f.v;
}

// B fragment: 32x16 (KxN) where storage is TRANSPOSED: Bt[n][k] (ldt = K stride).
// Layout used here: lane<16 hold K 0..15, lane>=16 hold K 16..31 (contiguous per lane).
DEVI v16bf frag_bt(const __bf16* baseT, int ldt, int k0, int n0, int lane) {
    BF16Frag f;
    const __bf16* row =
        baseT + (size_t)(n0 + (lane & 15)) * ldt + k0 + ((lane >> 4) << 4);
    f.q[0] = *(const uint4*)(row);
    f.q[1] = *(const uint4*)(row + 8);
    return f.v;
}

// B fragment: 32x16 (KxN) from row-major B[k][n] (strided; used for LDS V only).
DEVI v16bf frag_b(const __bf16* base, int ld, int k0, int n0, int lane) {
    BF16Frag f;
    const __bf16* p =
        base + (size_t)(k0 + ((lane >> 4) << 4)) * ld + n0 + (lane & 15);
#pragma unroll
    for (int j = 0; j < 8; ++j) {
        f.s[2 * j]     = p[(2 * j) * ld];
        f.s[2 * j + 1] = p[(2 * j + 1) * ld];
    }
    return f.v;
}

DEVI float wave_sum(float v) {
#pragma unroll
    for (int m = 16; m >= 1; m >>= 1) v += __shfl_xor(v, m, 32);
    return v;
}

// ---- CDNA5 async global->LDS (ASYNCcnt) ------------------------------------
// LDS byte address from a VGPR; global address = 64-bit SGPR base + 32-bit VGPR offset.
DEVI void async_copy_b128(unsigned lds_byte_addr, const void* sbase,
                          unsigned glb_byte_off) {
    asm volatile("global_load_async_to_lds_b128 %0, %1, %2"
                 :
                 : "v"(lds_byte_addr), "v"(glb_byte_off), "s"(sbase)
                 : "memory");
}
DEVI void wait_asynccnt0() {
    asm volatile("s_wait_asynccnt 0x0" ::: "memory");
}
DEVI unsigned lds_addr_of(const void* p) {       // low 32 bits = LDS offset
    return (unsigned)(uintptr_t)p;
}

// ---------------------------------------------------------------------------
// K0: convert + transpose all weights to bf16 (Wt[n][k], contiguous-K loads).
// ---------------------------------------------------------------------------
__global__ void __launch_bounds__(256)
swin_cvt_weights(const float* __restrict__ qkvw, const float* __restrict__ projw,
                 const float* __restrict__ w1,   const float* __restrict__ w2,
                 __bf16* __restrict__ qkvw_t, __bf16* __restrict__ projw_t,
                 __bf16* __restrict__ w1_t,   __bf16* __restrict__ w2_t) {
    int id = blockIdx.x * 256 + threadIdx.x;
    if (id < 196608) {                 // qkv_w [256][768] -> [768][256]
        int k = id / 768, n = id % 768;
        qkvw_t[n * 256 + k] = tobf(qkvw[id]);
    } else if (id < 262144) {          // proj_w [256][256] -> transposed
        int i = id - 196608; int k = i / 256, n = i % 256;
        projw_t[n * 256 + k] = tobf(projw[i]);
    } else if (id < 524288) {          // mlp_w1 [256][1024] -> [1024][256]
        int i = id - 262144; int k = i / 1024, n = i % 1024;
        w1_t[n * 256 + k] = tobf(w1[i]);
    } else if (id < 786432) {          // mlp_w2 [1024][256] -> [256][1024]
        int i = id - 524288; int k = i / 256, n = i % 256;
        w2_t[n * 1024 + k] = tobf(w2[i]);
    }
}

// ---------------------------------------------------------------------------
// K1: fused LN1 + cyclic shift + window partition + QKV GEMM (per window).
// block = 256 threads (8 waves), one window per block. LDS: 64x256 bf16 = 32KB.
// ---------------------------------------------------------------------------
__global__ void __launch_bounds__(256)
swin_ln1_qkv(const float* __restrict__ x, const __bf16* __restrict__ qkvw_t,
             const float* __restrict__ qkvb, const float* __restrict__ g,
             const float* __restrict__ bln,
             __bf16* __restrict__ wq, __bf16* __restrict__ wk,
             __bf16* __restrict__ wv) {
    extern __shared__ char smem[];
    __bf16* a_lds = (__bf16*)smem;          // [64][256]

    const int tid = threadIdx.x, lane = tid & 31, w = tid >> 5;
    const int wi = blockIdx.x;
    const int b = wi >> 6, w64 = wi & 63, wh = w64 >> 3, ww = w64 & 7;

    // one wave per token: LN over 256 dims, gather through roll(-3)+partition
    for (int t = w; t < 49; t += 8) {
        int r = t / 7, c = t % 7;
        int sh = (wh * 7 + r + 3) % 56;
        int sw = (ww * 7 + c + 3) % 56;
        const float* row = x + ((size_t)b * 3136 + sh * 56 + sw) * 256;
        float v[8]; float s = 0.f;
#pragma unroll
        for (int i = 0; i < 8; ++i) { v[i] = row[lane + 32 * i]; s += v[i]; }
        float mu = wave_sum(s) * (1.f / 256.f);
        float qv = 0.f;
#pragma unroll
        for (int i = 0; i < 8; ++i) { float d = v[i] - mu; qv += d * d; }
        float rstd = rsqrtf(wave_sum(qv) * (1.f / 256.f) + 1e-5f);
#pragma unroll
        for (int i = 0; i < 8; ++i) {
            int cc = lane + 32 * i;
            a_lds[t * 256 + cc] = tobf((v[i] - mu) * rstd * g[cc] + bln[cc]);
        }
    }
    for (int i = tid; i < 15 * 256; i += 256) a_lds[49 * 256 + i] = tobf(0.f);
    __syncthreads();

    // GEMM: [64x256] x [256x768]; wave w owns 96 output columns
    for (int nt = 0; nt < 6; ++nt) {
        int n0 = w * 96 + nt * 16;
        int ncol = n0 + (lane & 15);
        float bias = qkvb[ncol];
        for (int mt = 0; mt < 4; ++mt) {
            v8f acc = splat8(bias);
#pragma unroll
            for (int k0 = 0; k0 < 256; k0 += 32) {
                v16bf a  = frag_a(a_lds, 256, mt * 16, k0, lane);
                v16bf bf = frag_bt(qkvw_t, 256, k0, n0, lane);
                acc = wmma_bf16(a, bf, acc);
            }
            int part = ncol >> 8, d = ncol & 255;
            __bf16* dst =
                (part == 0 ? wq : part == 1 ? wk : wv) + (size_t)wi * 16384 + d;
            float scale = (part == 0) ? 0.17677669529663687f : 1.0f; // hd^-0.5
#pragma unroll
            for (int vv = 0; vv < 8; ++vv) {
                int m = mt * 16 + vv + ((lane >> 4) << 3);
                dst[m * 256] = tobf(acc[vv] * scale);
            }
        }
    }
}

// ---------------------------------------------------------------------------
// K2: windowed attention. One window per block, one head per wave.
// LDS: Q,K,V 3x32KB + P 8x(64x64) bf16 64KB = 160KB of the 320KB WGP LDS.
// Tiles pulled with GLOBAL_LOAD_ASYNC_TO_LDS_B128 (ASYNCcnt).
// ---------------------------------------------------------------------------
__global__ void __launch_bounds__(256)
swin_attn(const __bf16* __restrict__ wq, const __bf16* __restrict__ wk,
          const __bf16* __restrict__ wv, const float* __restrict__ btab,
          __bf16* __restrict__ aout) {
    extern __shared__ char smem[];
    __bf16* q_lds = (__bf16*)smem;   // [64][256]
    __bf16* k_lds = q_lds + 16384;
    __bf16* v_lds = k_lds + 16384;
    __bf16* p_lds = v_lds + 16384;   // [8][64][64]

    const int tid = threadIdx.x, lane = tid & 31, head = tid >> 5;
    const int wi = blockIdx.x;
    const int w64 = wi & 63, wh = w64 >> 3, ww = w64 & 7;

    // async copy Q,K,V tiles (32KB each) straight into LDS
    {
        const unsigned q0 = lds_addr_of(q_lds);
        const unsigned k0a = lds_addr_of(k_lds);
        const unsigned v0a = lds_addr_of(v_lds);
        const void* gq = (const void*)(wq + (size_t)wi * 16384);
        const void* gk = (const void*)(wk + (size_t)wi * 16384);
        const void* gv = (const void*)(wv + (size_t)wi * 16384);
        for (int ch = tid; ch < 2048; ch += 256) {
            unsigned boff = (unsigned)ch * 16u;
            async_copy_b128(q0 + boff, gq, boff);
            async_copy_b128(k0a + boff, gk, boff);
            async_copy_b128(v0a + boff, gv, boff);
        }
        wait_asynccnt0();
    }
    __syncthreads();

    // S = Q·Kt : 64x64, K=32 -> 16 single-shot WMMAs per head
    v8f acc[4][4];
#pragma unroll
    for (int mt = 0; mt < 4; ++mt) {
        v16bf a = frag_a(q_lds, 256, mt * 16, head * 32, lane);
#pragma unroll
        for (int nt = 0; nt < 4; ++nt) {
            v16bf bf = frag_bt(k_lds, 256, head * 32, nt * 16, lane); // Kt
            acc[mt][nt] = wmma_bf16(a, bf, splat8(0.f));
        }
    }

    // + relative position bias + shifted-window mask (computed inline)
#pragma unroll
    for (int mt = 0; mt < 4; ++mt)
#pragma unroll
        for (int nt = 0; nt < 4; ++nt)
#pragma unroll
            for (int vv = 0; vv < 8; ++vv) {
                int m = mt * 16 + vv + ((lane >> 4) << 3);
                int n = nt * 16 + (lane & 15);
                float s = acc[mt][nt][vv];
                if (m < 49 && n < 49) {
                    int rq = m / 7, cq = m % 7, rk = n / 7, ck = n % 7;
                    s += btab[((rq - rk + 6) * 13 + (cq - ck + 6)) * 8 + head];
                    int hq = wh * 7 + rq, wqp = ww * 7 + cq;
                    int hk = wh * 7 + rk, wkp = ww * 7 + ck;
                    int codeq = (hq < 49 ? 0 : hq < 53 ? 1 : 2) * 3 +
                                (wqp < 49 ? 0 : wqp < 53 ? 1 : 2);
                    int codek = (hk < 49 ? 0 : hk < 53 ? 1 : 2) * 3 +
                                (wkp < 49 ? 0 : wkp < 53 ? 1 : 2);
                    if (codeq != codek) s -= 100.f;
                } else {
                    s = -1e30f;   // pad columns/rows
                }
                acc[mt][nt][vv] = s;
            }

    // row softmax (row lives in a 16-lane half; reduce via shfl_xor 8..1)
    __bf16* ph = p_lds + head * 4096;
#pragma unroll
    for (int mt = 0; mt < 4; ++mt)
#pragma unroll
        for (int vv = 0; vv < 8; ++vv) {
            float mx = -1e30f;
#pragma unroll
            for (int nt = 0; nt < 4; ++nt) mx = fmaxf(mx, acc[mt][nt][vv]);
#pragma unroll
            for (int d = 8; d >= 1; d >>= 1) mx = fmaxf(mx, __shfl_xor(mx, d, 32));
            float sum = 0.f;
#pragma unroll
            for (int nt = 0; nt < 4; ++nt) {
                float e = __expf(acc[mt][nt][vv] - mx);
                acc[mt][nt][vv] = e; sum += e;
            }
#pragma unroll
            for (int d = 8; d >= 1; d >>= 1) sum += __shfl_xor(sum, d, 32);
            float inv = 1.f / sum;
            int m = mt * 16 + vv + ((lane >> 4) << 3);
            int n = lane & 15;
#pragma unroll
            for (int nt = 0; nt < 4; ++nt)
                ph[m * 64 + nt * 16 + n] = tobf(acc[mt][nt][vv] * inv);
        }
    __syncthreads();

    // O = P·V : 64x32, K=64 -> 16 WMMAs per head
#pragma unroll
    for (int mt = 0; mt < 4; ++mt)
#pragma unroll
        for (int nt = 0; nt < 2; ++nt) {
            v8f o = splat8(0.f);
#pragma unroll
            for (int ks = 0; ks < 2; ++ks) {
                v16bf a  = frag_a(ph, 64, mt * 16, ks * 32, lane);
                v16bf bf = frag_b(v_lds + head * 32, 256, ks * 32, nt * 16, lane);
                o = wmma_bf16(a, bf, o);
            }
#pragma unroll
            for (int vv = 0; vv < 8; ++vv) {
                int m = mt * 16 + vv + ((lane >> 4) << 3);
                int n = head * 32 + nt * 16 + (lane & 15);
                aout[(size_t)wi * 16384 + m * 256 + n] = tobf(o[vv]);
            }
        }
}

// ---------------------------------------------------------------------------
// K3: proj GEMM + window reverse + un-shift scatter + residual add -> x1 (f32)
// ---------------------------------------------------------------------------
__global__ void __launch_bounds__(256)
swin_proj(const __bf16* __restrict__ aout, const __bf16* __restrict__ projw_t,
          const float* __restrict__ pb, const float* __restrict__ x,
          float* __restrict__ x1) {
    extern __shared__ char smem[];
    __bf16* a_lds = (__bf16*)smem;   // [64][256]

    const int tid = threadIdx.x, lane = tid & 31, w = tid >> 5;
    const int wi = blockIdx.x;
    const int b = wi >> 6, w64 = wi & 63, wh = w64 >> 3, ww = w64 & 7;

    {   // async tile load (32KB)
        const unsigned a0 = lds_addr_of(a_lds);
        const void* ga = (const void*)(aout + (size_t)wi * 16384);
        for (int ch = tid; ch < 2048; ch += 256) {
            unsigned boff = (unsigned)ch * 16u;
            async_copy_b128(a0 + boff, ga, boff);
        }
        wait_asynccnt0();
    }
    __syncthreads();

    for (int nt = 0; nt < 2; ++nt) {
        int n0 = w * 32 + nt * 16;
        int ncol = n0 + (lane & 15);
        float bias = pb[ncol];
        for (int mt = 0; mt < 4; ++mt) {
            v8f acc = splat8(bias);
#pragma unroll
            for (int k0 = 0; k0 < 256; k0 += 32) {
                v16bf a  = frag_a(a_lds, 256, mt * 16, k0, lane);
                v16bf bf = frag_bt(projw_t, 256, k0, n0, lane);
                acc = wmma_bf16(a, bf, acc);
            }
#pragma unroll
            for (int vv = 0; vv < 8; ++vv) {
                int m = mt * 16 + vv + ((lane >> 4) << 3);
                if (m < 49) {
                    int r = m / 7, c = m % 7;
                    int dh = (wh * 7 + r + 3) % 56;   // reverse roll(+3)
                    int dw = (ww * 7 + c + 3) % 56;
                    size_t gi = ((size_t)b * 3136 + dh * 56 + dw) * 256 + ncol;
                    x1[gi] = x[gi] + acc[vv];
                }
            }
        }
    }
}

// ---------------------------------------------------------------------------
// K4: fused LN2 + fc1(WMMA) + GELU (hidden kept in LDS) + fc2(WMMA) + residual
// 64 tokens per block. LDS: 32KB (ln tile) + 128KB (64x1024 bf16 hidden).
// ---------------------------------------------------------------------------
__global__ void __launch_bounds__(256)
swin_mlp(const float* __restrict__ x1, const __bf16* __restrict__ w1_t,
         const float* __restrict__ b1, const __bf16* __restrict__ w2_t,
         const float* __restrict__ b2, const float* __restrict__ g,
         const float* __restrict__ bb, float* __restrict__ out) {
    extern __shared__ char smem[];
    __bf16* a_lds = (__bf16*)smem;   // [64][256]
    __bf16* h_lds = a_lds + 16384;   // [64][1024]

    const int tid = threadIdx.x, lane = tid & 31, w = tid >> 5;
    const size_t t0 = (size_t)blockIdx.x * 64;

    for (int lt = w; lt < 64; lt += 8) {
        const float* row = x1 + (t0 + lt) * 256;
        float v[8]; float s = 0.f;
#pragma unroll
        for (int i = 0; i < 8; ++i) { v[i] = row[lane + 32 * i]; s += v[i]; }
        float mu = wave_sum(s) * (1.f / 256.f);
        float qv = 0.f;
#pragma unroll
        for (int i = 0; i < 8; ++i) { float d = v[i] - mu; qv += d * d; }
        float rstd = rsqrtf(wave_sum(qv) * (1.f / 256.f) + 1e-5f);
#pragma unroll
        for (int i = 0; i < 8; ++i) {
            int cc = lane + 32 * i;
            a_lds[lt * 256 + cc] = tobf((v[i] - mu) * rstd * g[cc] + bb[cc]);
        }
    }
    __syncthreads();

    // fc1: [64x256]x[256x1024] + GELU -> LDS ; wave w owns 128 columns
    for (int nt = 0; nt < 8; ++nt) {
        int n0 = w * 128 + nt * 16;
        int ncol = n0 + (lane & 15);
        float bias = b1[ncol];
        for (int mt = 0; mt < 4; ++mt) {
            v8f acc = splat8(bias);
#pragma unroll
            for (int k0 = 0; k0 < 256; k0 += 32) {
                v16bf a  = frag_a(a_lds, 256, mt * 16, k0, lane);
                v16bf bf = frag_bt(w1_t, 256, k0, n0, lane);
                acc = wmma_bf16(a, bf, acc);
            }
#pragma unroll
            for (int vv = 0; vv < 8; ++vv) {
                int m = mt * 16 + vv + ((lane >> 4) << 3);
                float t = acc[vv];
                float ge = 0.5f * t * (1.f + erff(t * 0.70710678118654752f));
                h_lds[m * 1024 + ncol] = tobf(ge);
            }
        }
    }
    __syncthreads();

    // fc2: [64x1024]x[1024x256] + bias + residual -> out
    for (int nt = 0; nt < 2; ++nt) {
        int n0 = w * 32 + nt * 16;
        int ncol = n0 + (lane & 15);
        float bias = b2[ncol];
        for (int mt = 0; mt < 4; ++mt) {
            v8f acc = splat8(bias);
#pragma unroll
            for (int k0 = 0; k0 < 1024; k0 += 32) {
                v16bf a  = frag_a(h_lds, 1024, mt * 16, k0, lane);
                v16bf bf = frag_bt(w2_t, 1024, k0, n0, lane);
                acc = wmma_bf16(a, bf, acc);
            }
#pragma unroll
            for (int vv = 0; vv < 8; ++vv) {
                int m = mt * 16 + vv + ((lane >> 4) << 3);
                size_t gi = (t0 + m) * 256 + ncol;
                out[gi] = x1[gi] + acc[vv];
            }
        }
    }
}

// ---------------------------------------------------------------------------
// Host launcher
// ---------------------------------------------------------------------------
extern "C" void kernel_launch(void* const* d_in, const int* in_sizes, int n_in,
                              void* d_out, int out_size, void* d_ws, size_t ws_size,
                              hipStream_t stream) {
    (void)in_sizes; (void)n_in; (void)out_size; (void)ws_size;

    const float* x        = (const float*)d_in[0];
    const float* qkv_w    = (const float*)d_in[1];
    const float* qkv_b    = (const float*)d_in[2];
    const float* proj_w   = (const float*)d_in[3];
    const float* proj_b   = (const float*)d_in[4];
    const float* ln1_g    = (const float*)d_in[5];
    const float* ln1_b    = (const float*)d_in[6];
    const float* ln2_g    = (const float*)d_in[7];
    const float* ln2_b    = (const float*)d_in[8];
    const float* mlp_w1   = (const float*)d_in[9];
    const float* mlp_b1   = (const float*)d_in[10];
    const float* mlp_w2   = (const float*)d_in[11];
    const float* mlp_b2   = (const float*)d_in[12];
    const float* btab     = (const float*)d_in[13];
    float* out            = (float*)d_out;

    // workspace layout (bytes)
    constexpr size_t SZ_WIN  = (size_t)2048 * 64 * 256 * 2;   // 64 MiB (bf16)
    constexpr size_t OFF_WQ  = 0;
    constexpr size_t OFF_WK  = OFF_WQ + SZ_WIN;
    constexpr size_t OFF_WV  = OFF_WK + SZ_WIN;
    constexpr size_t OFF_AO  = OFF_WV + SZ_WIN;
    constexpr size_t OFF_X1  = OFF_AO + SZ_WIN;
    constexpr size_t OFF_QW  = OFF_X1 + (size_t)100352 * 256 * 4;
    constexpr size_t OFF_PW  = OFF_QW + (size_t)768 * 256 * 2;
    constexpr size_t OFF_W1  = OFF_PW + (size_t)256 * 256 * 2;
    constexpr size_t OFF_W2  = OFF_W1 + (size_t)1024 * 256 * 2;

    char* ws = (char*)d_ws;
    __bf16* wq      = (__bf16*)(ws + OFF_WQ);
    __bf16* wk      = (__bf16*)(ws + OFF_WK);
    __bf16* wv      = (__bf16*)(ws + OFF_WV);
    __bf16* aout    = (__bf16*)(ws + OFF_AO);
    float*  x1      = (float*)(ws + OFF_X1);
    __bf16* qkvw_t  = (__bf16*)(ws + OFF_QW);
    __bf16* projw_t = (__bf16*)(ws + OFF_PW);
    __bf16* w1_t    = (__bf16*)(ws + OFF_W1);
    __bf16* w2_t    = (__bf16*)(ws + OFF_W2);

    // allow 160KB dynamic LDS (WGP has 320KB); ignore errors (no-op on ROCm)
    (void)hipFuncSetAttribute((const void*)swin_attn,
                              hipFuncAttributeMaxDynamicSharedMemorySize, 163840);
    (void)hipFuncSetAttribute((const void*)swin_mlp,
                              hipFuncAttributeMaxDynamicSharedMemorySize, 163840);

    swin_cvt_weights<<<dim3(3072), dim3(256), 0, stream>>>(
        qkv_w, proj_w, mlp_w1, mlp_w2, qkvw_t, projw_t, w1_t, w2_t);

    swin_ln1_qkv<<<dim3(2048), dim3(256), 32768, stream>>>(
        x, qkvw_t, qkv_b, ln1_g, ln1_b, wq, wk, wv);

    swin_attn<<<dim3(2048), dim3(256), 163840, stream>>>(
        wq, wk, wv, btab, aout);

    swin_proj<<<dim3(2048), dim3(256), 32768, stream>>>(
        aout, projw_t, proj_b, x, x1);

    swin_mlp<<<dim3(1568), dim3(256), 163840, stream>>>(
        x1, w1_t, mlp_b1, w2_t, mlp_b2, ln2_g, ln2_b, out);
}
